// GAT_54554674593969
// MI455X (gfx1250) — matrix operand order repaired and checked by reading, not compiled, not run on hardware
//
#include <hip/hip_runtime.h>
#include <hip/hip_bf16.h>

// ---------------- problem constants ----------------
#define NN      16384
#define EE      262144
#define GG      32
#define NPG     512
#define DFEAT   256
#define AADIM   64
#define DIN0    320     // AADIM + DFEAT
#define HO0     128     // H0*O0
#define O0v     64
#define DIN1    128
#define HO1     512     // H1*O1
#define O1v     256
#define NEGS    0.2f

typedef _Float16 half8  __attribute__((ext_vector_type(8)));
typedef _Float16 half16 __attribute__((ext_vector_type(16)));
typedef float    float8 __attribute__((ext_vector_type(8)));

// ---------------- workspace layout (bytes) ----------------
// Region A: phase-1 buffers, later reused for ft1 and h2 (lifetimes disjoint).
static constexpr size_t OFF_H16   = 0;                                    // N*320*2  = 10485760
static constexpr size_t OFF_FT0   = OFF_H16   + (size_t)NN*DIN0*2;        // N*128*4  = 8388608
static constexpr size_t OFF_HRES0 = OFF_FT0   + (size_t)NN*HO0*4;
static constexpr size_t OFF_RST0  = OFF_HRES0 + (size_t)NN*HO0*4;
static constexpr size_t REGION_A_END = OFF_RST0 + (size_t)NN*HO0*4;       // 35,651,584
static constexpr size_t OFF_FT1   = 0;   // 33,554,432  (overlays region A, after h1 built)
static constexpr size_t OFF_H2    = 0;   // 16,777,216  (overlays region A, after ft1 consumed)

static constexpr size_t OFF_HRES1 = REGION_A_END;                          // N*512*4 = 33554432
static constexpr size_t OFF_RST1  = OFF_HRES1 + (size_t)NN*HO1*4;
static constexpr size_t OFF_H116  = OFF_RST1  + (size_t)NN*HO1*4;          // N*128*2 = 4194304
static constexpr size_t OFF_EL0   = OFF_H116  + (size_t)NN*DIN1*2;
static constexpr size_t OFF_ER0   = OFF_EL0   + (size_t)NN*2*4;
static constexpr size_t OFF_EMAX0 = OFF_ER0   + (size_t)NN*2*4;
static constexpr size_t OFF_DEN0  = OFF_EMAX0 + (size_t)NN*2*4;
static constexpr size_t OFF_EZ0   = OFF_DEN0  + (size_t)NN*2*4;            // E*2*4 = 2097152
static constexpr size_t OFF_EL1   = OFF_EZ0   + (size_t)EE*2*4;
static constexpr size_t OFF_ER1   = OFF_EL1   + (size_t)NN*2*4;
static constexpr size_t OFF_EMAX1 = OFF_ER1   + (size_t)NN*2*4;
static constexpr size_t OFF_DEN1  = OFF_EMAX1 + (size_t)NN*2*4;
static constexpr size_t OFF_EZ1   = OFF_DEN1  + (size_t)NN*2*4;
static constexpr size_t OFF_HG    = OFF_EZ1   + (size_t)EE*2*4;            // 32*256*4
static constexpr size_t OFF_X1    = OFF_HG    + (size_t)GG*O1v*4;          // 32*128*4
static constexpr size_t OFF_X2    = OFF_X1    + (size_t)GG*128*4;          // 32*64*4
static constexpr size_t OFF_W0T   = OFF_X2    + (size_t)GG*64*4;           // 128*320*2
static constexpr size_t OFF_R0T   = OFF_W0T   + (size_t)HO0*DIN0*2;
static constexpr size_t OFF_W1T   = OFF_R0T   + (size_t)HO0*DIN0*2;        // 512*128*2
static constexpr size_t OFF_R1T   = OFF_W1T   + (size_t)HO1*DIN1*2;

// ---------------- helpers ----------------
__device__ __forceinline__ float eluf(float x) {
    return x > 0.0f ? x : (__expf(x) - 1.0f);
}
__device__ __forceinline__ float lrelu(float x) {
    return x > 0.0f ? x : NEGS * x;
}
// sign-aware float atomic max (works with -FLT_MAX init)
__device__ __forceinline__ void atomicMaxFloat(float* addr, float val) {
    if (val >= 0.0f) {
        atomicMax((int*)addr, __float_as_int(val));
    } else {
        atomicMin((unsigned int*)addr, (unsigned int)__float_as_int(val));
    }
}

// ---------------- kernels ----------------
__global__ void fill_kernel(float* __restrict__ p, float v, int n) {
    int i = blockIdx.x * blockDim.x + threadIdx.x;
    if (i < n) p[i] = v;
}

__global__ void fill4_kernel(float4* __restrict__ p, float v, int n4) {
    int i = blockIdx.x * blockDim.x + threadIdx.x;
    if (i < n4) p[i] = make_float4(v, v, v, v);
}

// h = concat(aa_emb[ref]*aa_emb[alt[gid]], feat), stored as f16 [N,320]
__global__ void build_h16_kernel(const float* __restrict__ feat,
                                 const float* __restrict__ aa_emb,
                                 const int* __restrict__ ref_aa,
                                 const int* __restrict__ alt_aa,
                                 const int* __restrict__ graph_id,
                                 _Float16* __restrict__ h16) {
    int idx = blockIdx.x * blockDim.x + threadIdx.x;
    if (idx >= NN * DIN0) return;
    int n = idx / DIN0, d = idx - n * DIN0;
    float v;
    if (d < AADIM) {
        int r = ref_aa[n];
        int a = alt_aa[graph_id[n]];
        v = aa_emb[r * AADIM + d] * aa_emb[a * AADIM + d];
    } else {
        v = feat[(size_t)n * DFEAT + (d - AADIM)];
    }
    h16[idx] = (_Float16)v;
}

// f32 W[din,dout] -> f16 Wt[dout,din] (transposed, contiguous-K rows)
__global__ void w_to_f16t_kernel(const float* __restrict__ W, _Float16* __restrict__ Wt,
                                 int din, int dout) {
    int idx = blockIdx.x * blockDim.x + threadIdx.x;
    if (idx >= din * dout) return;
    int k = idx / dout, j = idx - k * dout;
    Wt[(size_t)j * din + k] = (_Float16)W[idx];
}

// C[M,dout] = A16[M,din] @ Wt16[dout,din]^T (+bias), one 16x16 tile per wave,
// K in steps of 32 via v_wmma_f32_16x16x32_f16. Requires M%64==0, dout%16==0.
__global__ __launch_bounds__(128)
void wmma_gemm_kernel(const _Float16* __restrict__ A, const _Float16* __restrict__ Wt,
                      const float* __restrict__ bias, float* __restrict__ C,
                      int din, int dout) {
    const int tm   = (blockIdx.x * 4 + threadIdx.y) * 16;   // row tile
    const int tn   = blockIdx.y * 16;                       // col tile
    const int lane = threadIdx.x;                           // 0..31 (wave32)
    const int hi   = lane >> 4;                             // lane half
    const int l    = lane & 15;

    const _Float16* arow = A  + (size_t)(tm + l) * din;
    const _Float16* brow = Wt + (size_t)(tn + l) * din;

    float8 acc = {};
    for (int k = 0; k < din; k += 32) {
        union { half16 v; half8 h[2]; } ua, ub;
        // A 16x32 fragment: lane<16 -> K {k..k+7, k+16..k+23}; lane>=16 -> +8
        ua.h[0] = *(const half8*)(arow + k + hi * 8);
        ua.h[1] = *(const half8*)(arow + k + 16 + hi * 8);
        // B 32x16 fragment: column tn+l, same K order as A (lane<16 -> K {k..k+7, k+16..k+23}; lane>=16 -> +8)
        ub.h[0] = *(const half8*)(brow + k + hi * 8);
        ub.h[1] = *(const half8*)(brow + k + 16 + hi * 8);
        acc = __builtin_amdgcn_wmma_f32_16x16x32_f16(
            false, ua.v, false, ub.v, (short)0, acc, false, false);
    }
    const int col = tn + l;
    const float b = bias ? bias[col] : 0.0f;
    float* crow = C + (size_t)(tm + hi * 8) * dout + col;
#pragma unroll
    for (int r = 0; r < 8; ++r)
        crow[(size_t)r * dout] = acc[r] + b;
}

// el/er: per (node, head) dot(ft[n,h,:], a[h,:])
__global__ void attn_coef_kernel(const float* __restrict__ ft, const float* __restrict__ a,
                                 float* __restrict__ out, int O) {
    int idx = blockIdx.x * blockDim.x + threadIdx.x;
    if (idx >= NN * 2) return;
    int n = idx >> 1, h = idx & 1;
    const float* f  = ft + (size_t)n * 2 * O + (size_t)h * O;
    const float* av = a + (size_t)h * O;
    float s = 0.0f;
    for (int o = 0; o < O; ++o) s += f[o] * av[o];
    out[idx] = s;
}

__global__ void edge_max_kernel(const float* __restrict__ el, const float* __restrict__ er,
                                const int* __restrict__ src, const int* __restrict__ dst,
                                float* __restrict__ emax) {
    int idx = blockIdx.x * blockDim.x + threadIdx.x;
    if (idx >= EE * 2) return;
    int e = idx >> 1, h = idx & 1;
    int s = src[e], d = dst[e];
    float w = lrelu(el[s * 2 + h] + er[d * 2 + h]);
    atomicMaxFloat(&emax[d * 2 + h], w);
}

__global__ void edge_expsum_kernel(const float* __restrict__ el, const float* __restrict__ er,
                                   const float* __restrict__ emax,
                                   const int* __restrict__ src, const int* __restrict__ dst,
                                   float* __restrict__ ez, float* __restrict__ denom) {
    int idx = blockIdx.x * blockDim.x + threadIdx.x;
    if (idx >= EE * 2) return;
    int e = idx >> 1, h = idx & 1;
    int s = src[e], d = dst[e];
    float w = lrelu(el[s * 2 + h] + er[d * 2 + h]);
    float z = __expf(w - emax[d * 2 + h]);
    ez[idx] = z;
    atomicAdd(&denom[d * 2 + h], z);
}

// alpha[e,h] = ez[e,h] / denom[dst[e],h]  (in-place over ez)
__global__ void edge_alpha_kernel(float* __restrict__ ez, const float* __restrict__ denom,
                                  const int* __restrict__ dst) {
    int idx = blockIdx.x * blockDim.x + threadIdx.x;
    if (idx >= EE * 2) return;
    int e = idx >> 1, h = idx & 1;
    ez[idx] = ez[idx] / denom[dst[e] * 2 + h];
}

// rst[dst, f..f+3] += ft[src, f..f+3] * alpha(e, head(f)); float4 per thread
__global__ void edge_agg4_kernel(const float4* __restrict__ ft4, const float* __restrict__ alpha,
                                 const int* __restrict__ src, const int* __restrict__ dst,
                                 float* __restrict__ rst, int HO, int O) {
    long long idx = (long long)blockIdx.x * blockDim.x + threadIdx.x;
    const int HO4 = HO >> 2;
    if (idx >= (long long)EE * HO4) return;
    int e = (int)(idx / HO4);
    int q = (int)(idx - (long long)e * HO4);   // quad index within the row
    int f = q << 2;
    int h = f / O;                              // head is uniform across the quad (O % 4 == 0)
    int s = src[e], d = dst[e];
    float a = alpha[e * 2 + h];
    float4 v = ft4[(size_t)s * HO4 + q];        // global_load_b128 gather
    float* r = rst + (size_t)d * HO + f;
    atomicAdd(r + 0, v.x * a);
    atomicAdd(r + 1, v.y * a);
    atomicAdd(r + 2, v.z * a);
    atomicAdd(r + 3, v.w * a);
}

// h1 = elu(rst0 + (h@res0 + b0)) -> f16
__global__ void layer0_finish_kernel(const float* __restrict__ rst, const float* __restrict__ hres,
                                     _Float16* __restrict__ h1) {
    int idx = blockIdx.x * blockDim.x + threadIdx.x;
    if (idx >= NN * HO0) return;
    h1[idx] = (_Float16)eluf(rst[idx] + hres[idx]);
}

// h2 = elu(mean over heads of (rst1 + hres1))
__global__ void layer1_finish_kernel(const float* __restrict__ rst, const float* __restrict__ hres,
                                     float* __restrict__ h2) {
    int idx = blockIdx.x * blockDim.x + threadIdx.x;
    if (idx >= NN * O1v) return;
    int n = idx / O1v, o = idx - n * O1v;
    size_t base = (size_t)n * HO1;
    float t0 = rst[base + o] + hres[base + o];
    float t1 = rst[base + O1v + o] + hres[base + O1v + o];
    h2[idx] = eluf(0.5f * (t0 + t1));
}

// per-graph mean over the 512 contiguous nodes of each graph
__global__ void readout_kernel(const float* __restrict__ h2, float* __restrict__ hg) {
    int g = blockIdx.x;          // 32
    int o = threadIdx.x;         // 256
    float s = 0.0f;
    const float* base = h2 + ((size_t)g * NPG) * O1v + o;
    for (int i = 0; i < NPG; ++i) s += base[(size_t)i * O1v];
    hg[(size_t)g * O1v + o] = s * (1.0f / NPG);
}

// tiny dense layer: y[g,j] = act(sum_k x[g,k]*W[k,j] + b[j])
__global__ void mlp_kernel(const float* __restrict__ x, const float* __restrict__ W,
                           const float* __restrict__ b, float* __restrict__ y,
                           int din, int dout, int do_relu) {
    int g = blockIdx.x;
    int j = threadIdx.x;
    if (j >= dout) return;
    float s = b[j];
    const float* xr = x + (size_t)g * din;
    for (int k = 0; k < din; ++k) s += xr[k] * W[(size_t)k * dout + j];
    y[(size_t)g * dout + j] = do_relu ? fmaxf(s, 0.0f) : s;
}

// ---------------- launcher ----------------
extern "C" void kernel_launch(void* const* d_in, const int* in_sizes, int n_in,
                              void* d_out, int out_size, void* d_ws, size_t ws_size,
                              hipStream_t stream) {
    const float* feat   = (const float*)d_in[0];
    const float* aa_emb = (const float*)d_in[1];
    const float* W0     = (const float*)d_in[2];
    const float* al0    = (const float*)d_in[3];
    const float* ar0    = (const float*)d_in[4];
    const float* res0   = (const float*)d_in[5];
    const float* b0     = (const float*)d_in[6];
    const float* W1     = (const float*)d_in[7];
    const float* al1    = (const float*)d_in[8];
    const float* ar1    = (const float*)d_in[9];
    const float* res1   = (const float*)d_in[10];
    const float* b1     = (const float*)d_in[11];
    const float* mW0    = (const float*)d_in[12];
    const float* mb0    = (const float*)d_in[13];
    const float* mW1    = (const float*)d_in[14];
    const float* mb1    = (const float*)d_in[15];
    const float* mW2    = (const float*)d_in[16];
    const float* mb2    = (const float*)d_in[17];
    const int* ref_aa   = (const int*)d_in[18];
    const int* alt_aa   = (const int*)d_in[19];
    const int* src      = (const int*)d_in[20];
    const int* dst      = (const int*)d_in[21];
    const int* graph_id = (const int*)d_in[22];

    char* ws = (char*)d_ws;
    _Float16* h16   = (_Float16*)(ws + OFF_H16);
    float*    ft0   = (float*)(ws + OFF_FT0);
    float*    hres0 = (float*)(ws + OFF_HRES0);
    float*    rst0  = (float*)(ws + OFF_RST0);
    float*    ft1   = (float*)(ws + OFF_FT1);
    float*    h2    = (float*)(ws + OFF_H2);
    float*    hres1 = (float*)(ws + OFF_HRES1);
    float*    rst1  = (float*)(ws + OFF_RST1);
    _Float16* h116  = (_Float16*)(ws + OFF_H116);
    float* el0 = (float*)(ws + OFF_EL0),  * er0 = (float*)(ws + OFF_ER0);
    float* em0 = (float*)(ws + OFF_EMAX0),* dn0 = (float*)(ws + OFF_DEN0);
    float* ez0 = (float*)(ws + OFF_EZ0);
    float* el1 = (float*)(ws + OFF_EL1),  * er1 = (float*)(ws + OFF_ER1);
    float* em1 = (float*)(ws + OFF_EMAX1),* dn1 = (float*)(ws + OFF_DEN1);
    float* ez1 = (float*)(ws + OFF_EZ1);
    float* hg  = (float*)(ws + OFF_HG);
    float* x1  = (float*)(ws + OFF_X1);
    float* x2  = (float*)(ws + OFF_X2);
    _Float16* w0t = (_Float16*)(ws + OFF_W0T);
    _Float16* r0t = (_Float16*)(ws + OFF_R0T);
    _Float16* w1t = (_Float16*)(ws + OFF_W1T);
    _Float16* r1t = (_Float16*)(ws + OFF_R1T);

    const int TPB = 256;
    #define NBLK(n) (((n) + TPB - 1) / TPB)

    // weight conversion (f32 -> f16, transposed)
    w_to_f16t_kernel<<<NBLK(DIN0*HO0), TPB, 0, stream>>>(W0,   w0t, DIN0, HO0);
    w_to_f16t_kernel<<<NBLK(DIN0*HO0), TPB, 0, stream>>>(res0, r0t, DIN0, HO0);
    w_to_f16t_kernel<<<NBLK(DIN1*HO1), TPB, 0, stream>>>(W1,   w1t, DIN1, HO1);
    w_to_f16t_kernel<<<NBLK(DIN1*HO1), TPB, 0, stream>>>(res1, r1t, DIN1, HO1);

    // node input features
    build_h16_kernel<<<NBLK(NN*DIN0), TPB, 0, stream>>>(feat, aa_emb, ref_aa, alt_aa,
                                                        graph_id, h16);

    // ---- layer 0 ----
    {
        dim3 blk(32, 4);
        dim3 grd(NN / 64, HO0 / 16);
        wmma_gemm_kernel<<<grd, blk, 0, stream>>>(h16, w0t, nullptr, ft0, DIN0, HO0);
        wmma_gemm_kernel<<<grd, blk, 0, stream>>>(h16, r0t, b0,      hres0, DIN0, HO0);
    }
    attn_coef_kernel<<<NBLK(NN*2), TPB, 0, stream>>>(ft0, al0, el0, O0v);
    attn_coef_kernel<<<NBLK(NN*2), TPB, 0, stream>>>(ft0, ar0, er0, O0v);
    fill_kernel<<<NBLK(NN*2), TPB, 0, stream>>>(em0, -3.402823e38f, NN*2);
    fill_kernel<<<NBLK(NN*2), TPB, 0, stream>>>(dn0, 0.0f, NN*2);
    fill4_kernel<<<NBLK(NN*HO0/4), TPB, 0, stream>>>((float4*)rst0, 0.0f, NN*HO0/4);
    edge_max_kernel   <<<NBLK(EE*2), TPB, 0, stream>>>(el0, er0, src, dst, em0);
    edge_expsum_kernel<<<NBLK(EE*2), TPB, 0, stream>>>(el0, er0, em0, src, dst, ez0, dn0);
    edge_alpha_kernel <<<NBLK(EE*2), TPB, 0, stream>>>(ez0, dn0, dst);
    {
        long long tot = (long long)EE * (HO0 / 4);
        edge_agg4_kernel<<<(unsigned)((tot + TPB - 1) / TPB), TPB, 0, stream>>>(
            (const float4*)ft0, ez0, src, dst, rst0, HO0, O0v);
    }
    layer0_finish_kernel<<<NBLK(NN*HO0), TPB, 0, stream>>>(rst0, hres0, h116);

    // ---- layer 1 ---- (ft1/h2 overlay region A; h16/ft0/hres0/rst0 are dead)
    {
        dim3 blk(32, 4);
        dim3 grd(NN / 64, HO1 / 16);
        wmma_gemm_kernel<<<grd, blk, 0, stream>>>(h116, w1t, nullptr, ft1, DIN1, HO1);
        wmma_gemm_kernel<<<grd, blk, 0, stream>>>(h116, r1t, b1,      hres1, DIN1, HO1);
    }
    attn_coef_kernel<<<NBLK(NN*2), TPB, 0, stream>>>(ft1, al1, el1, O1v);
    attn_coef_kernel<<<NBLK(NN*2), TPB, 0, stream>>>(ft1, ar1, er1, O1v);
    fill_kernel<<<NBLK(NN*2), TPB, 0, stream>>>(em1, -3.402823e38f, NN*2);
    fill_kernel<<<NBLK(NN*2), TPB, 0, stream>>>(dn1, 0.0f, NN*2);
    fill4_kernel<<<NBLK(NN*HO1/4), TPB, 0, stream>>>((float4*)rst1, 0.0f, NN*HO1/4);
    edge_max_kernel   <<<NBLK(EE*2), TPB, 0, stream>>>(el1, er1, src, dst, em1);
    edge_expsum_kernel<<<NBLK(EE*2), TPB, 0, stream>>>(el1, er1, em1, src, dst, ez1, dn1);
    edge_alpha_kernel <<<NBLK(EE*2), TPB, 0, stream>>>(ez1, dn1, dst);
    {
        long long tot = (long long)EE * (HO1 / 4);
        edge_agg4_kernel<<<(unsigned)((tot + TPB - 1) / TPB), TPB, 0, stream>>>(
            (const float4*)ft1, ez1, src, dst, rst1, HO1, O1v);
    }
    layer1_finish_kernel<<<NBLK(NN*O1v), TPB, 0, stream>>>(rst1, hres1, h2);

    // ---- readout + MLP ----
    readout_kernel<<<GG, O1v, 0, stream>>>(h2, hg);
    mlp_kernel<<<GG, 128, 0, stream>>>(hg, mW0, mb0, x1, O1v, 128, 1);
    mlp_kernel<<<GG, 128, 0, stream>>>(x1, mW1, mb1, x2, 128, 64, 1);
    mlp_kernel<<<GG, 32,  0, stream>>>(x2, mW2, mb2, (float*)d_out, 64, 2, 0);
}
